// MultiHeadAttn_3350074491188
// MI455X (gfx1250) — matrix operand order repaired and checked by reading, not compiled, not run on hardware
//
#include <hip/hip_runtime.h>

#define BSZ    4
#define SEQ    2048
#define DMODEL 512
#define NHEAD  8
#define HDIM   64
#define MROWS  (BSZ * SEQ)   // 8192

typedef __attribute__((ext_vector_type(16))) __bf16 v16bf;
typedef __attribute__((ext_vector_type(8)))  __bf16 v8bf;
typedef __attribute__((ext_vector_type(8)))  float  v8f;
typedef __attribute__((ext_vector_type(4)))  unsigned int v4u;
typedef __attribute__((ext_vector_type(8)))  int v8i;
typedef __attribute__((ext_vector_type(4)))  int v4i;

__device__ __forceinline__ v8f wmma_bf16(v16bf a, v16bf b, v8f c) {
  return __builtin_amdgcn_wmma_f32_16x16x32_bf16(
      /*neg_a=*/false, a, /*neg_b=*/false, b,
      /*c_mod=*/(short)0, c, /*reuse_a=*/false, /*reuse_b=*/false);
}

// ---------------------------------------------------------------------------
// TDM: async 2-D bf16 tile load Global -> LDS (CDNA5 ISA)
// group0: [0]=count=1, [1]=lds byte addr, [2..3]=global addr (57b) + type=2
// group1: data_size=1(2B) @bits17:16; tensor_dim0 @79:48; tensor_dim1 @111:80;
//         tile_dim0 @127:112; tile_dim1 @143:128; dim0_stride @207:160
// This toolchain exposes the 6-arg builtin (groups 0..3 + extra group + cpol).
// ---------------------------------------------------------------------------
__device__ __forceinline__ void tdm_load_2d(unsigned lds_addr, const void* gptr,
                                            unsigned tile_d0, unsigned tile_d1,
                                            unsigned tensor_d0, unsigned tensor_d1,
                                            unsigned stride0) {
  unsigned long long ga = (unsigned long long)gptr;
  v4u g0;
  g0[0] = 1u;                                                  // count=1
  g0[1] = lds_addr;
  g0[2] = (unsigned)ga;
  g0[3] = (unsigned)((ga >> 32) & 0x01FFFFFFu) | (2u << 30);   // type=2 (image)
  v8i g1;
  g1[0] = (int)(1u << 16);                                     // data_size = 2B
  g1[1] = (int)((tensor_d0 & 0xFFFFu) << 16);                  // tdim0[15:0]
  g1[2] = (int)(((tensor_d0 >> 16) & 0xFFFFu) | ((tensor_d1 & 0xFFFFu) << 16));
  g1[3] = (int)(((tensor_d1 >> 16) & 0xFFFFu) | (tile_d0 << 16));
  g1[4] = (int)(tile_d1 & 0xFFFFu);                            // tile_dim2 = 0
  g1[5] = (int)stride0;                                        // dim0 stride lo
  g1[6] = 0;
  g1[7] = 0;
  v4i z4 = {0, 0, 0, 0};
  v8i z8 = {0, 0, 0, 0, 0, 0, 0, 0};
  __builtin_amdgcn_tensor_load_to_lds(g0, g1, z4, z4, z8, 0);
}

// ---------------------------------------------------------------------------
// GEMM: Out[m,n] = sum_k A[m,k] * W[n,k]   (x @ W^T), one wave -> 32x64 tile.
// SMODE 0: bf16 out [b,h,s,d] | 1: bf16 out [b,h,d,s] | 2: fp32 out [m,n]
// ---------------------------------------------------------------------------
template <bool A_F32, int SMODE>
__global__ __launch_bounds__(32) void proj_gemm(const void*  __restrict__ Aarg,
                                                const float* __restrict__ W,
                                                void*        __restrict__ Out) {
  const int lane = threadIdx.x & 31;
  const int half = lane >> 4;   // 0: lanes 0-15, 1: lanes 16-31
  const int ln   = lane & 15;
  const int m0   = blockIdx.x * 32;
  const int n0   = blockIdx.y * 64;

  v8f acc[2][4] = {};

  for (int kb = 0; kb < DMODEL; kb += 32) {
    // ---- two A fragments (rows m0..m0+15, m0+16..m0+31) ----
    v16bf a[2];
#pragma unroll
    for (int mi = 0; mi < 2; ++mi) {
      const int arow = m0 + mi * 16 + ln;
      if (A_F32) {
        const float* ap = (const float*)Aarg + (size_t)arow * DMODEL + kb + half * 8;
        v8f f0 = *(const v8f*)ap;
        v8f f1 = *(const v8f*)(ap + 16);
#pragma unroll
        for (int i = 0; i < 8; ++i) { a[mi][i] = (__bf16)f0[i]; a[mi][i + 8] = (__bf16)f1[i]; }
      } else {
        const __bf16* ap = (const __bf16*)Aarg + (size_t)arow * DMODEL + kb + half * 8;
        v8bf h0 = *(const v8bf*)ap;
        v8bf h1 = *(const v8bf*)(ap + 16);
#pragma unroll
        for (int i = 0; i < 8; ++i) { a[mi][i] = h0[i]; a[mi][i + 8] = h1[i]; }
      }
    }
    // ---- 4 B fragments, each reused by both A fragments: 8 WMMAs ----
#pragma unroll
    for (int t = 0; t < 4; ++t) {
      const float* wp = W + (size_t)(n0 + t * 16 + ln) * DMODEL + kb + half * 8;
      v8f g0 = *(const v8f*)wp;
      v8f g1 = *(const v8f*)(wp + 16);
      v16bf bf;
#pragma unroll
      for (int i = 0; i < 8; ++i) { bf[i] = (__bf16)g0[i]; bf[i + 8] = (__bf16)g1[i]; }
      acc[0][t] = wmma_bf16(a[0], bf, acc[0][t]);
      acc[1][t] = wmma_bf16(a[1], bf, acc[1][t]);
    }
  }

  // ---- store: VGPR r <-> row (r + half*8), column = lane&15 ----
#pragma unroll
  for (int mi = 0; mi < 2; ++mi) {
#pragma unroll
    for (int t = 0; t < 4; ++t) {
      const int n = n0 + t * 16 + ln;
      const int h = n >> 6, d = n & 63;
#pragma unroll
      for (int r = 0; r < 8; ++r) {
        const int m = m0 + mi * 16 + half * 8 + r;
        const int b = m >> 11;          // / SEQ
        const int s = m & (SEQ - 1);
        if (SMODE == 0) {
          ((__bf16*)Out)[(((size_t)(b * NHEAD + h) * SEQ) + s) * HDIM + d] = (__bf16)acc[mi][t][r];
        } else if (SMODE == 1) {
          ((__bf16*)Out)[(((size_t)(b * NHEAD + h) * HDIM) + d) * SEQ + s] = (__bf16)acc[mi][t][r];
        } else {
          ((float*)Out)[(size_t)m * DMODEL + n] = acc[mi][t][r];
        }
      }
    }
  }
}

// ---------------------------------------------------------------------------
// Flash attention: one wave (block=32) owns a 16-row Q tile for one (b,h).
// K/V 32-key tiles are DMA'd into double-buffered LDS by the Tensor Data
// Mover; s_wait_tensorcnt(2) overlaps the DMA of block j+1 with the 8 WMMAs
// of block j. headV is pre-transposed [b,h,d,s] so tiles are dense 2-D.
// ---------------------------------------------------------------------------
__global__ __launch_bounds__(32) void flash_attn(const __bf16* __restrict__ hq,
                                                 const __bf16* __restrict__ hk,
                                                 const __bf16* __restrict__ hvt,
                                                 const int*    __restrict__ amask,
                                                 __bf16*       __restrict__ av) {
  __shared__ __bf16 bufk[2][32 * HDIM];   // K tile: 32 keys x 64 d   (4 KB each)
  __shared__ __bf16 bufv[2][HDIM * 32];   // V tile: 64 d x 32 keys   (4 KB each)
  __shared__ __bf16 lds_p[16 * 32];       // P tile (C-layout -> A-layout)

  const int lane = threadIdx.x & 31;
  const int half = lane >> 4;
  const int ln   = lane & 15;
  const int bh   = blockIdx.x;
  const int b    = bh >> 3;              // / NHEAD
  const int h    = bh & 7;
  const int m0   = blockIdx.y * 16;

  const __bf16* Qb = hq  + (size_t)bh * SEQ * HDIM;
  const __bf16* Kb = hk  + (size_t)bh * SEQ * HDIM;
  const __bf16* Vb = hvt + (size_t)bh * HDIM * SEQ;
  const int*    mk = amask + b * SEQ;

  // Q fragments for d = 0..31 and 32..63, kept in registers for the whole pass
  v16bf qa[2];
#pragma unroll
  for (int kk = 0; kk < 2; ++kk) {
    const __bf16* qp = Qb + (size_t)(m0 + ln) * HDIM + kk * 32 + half * 16;
    v8bf q0 = *(const v8bf*)qp;
    v8bf q1 = *(const v8bf*)(qp + 8);
#pragma unroll
    for (int i = 0; i < 8; ++i) { qa[kk][i] = q0[i]; qa[kk][i + 8] = q1[i]; }
  }

  v8f o[4] = {};
  v8f mrow, lrow;
#pragma unroll
  for (int r = 0; r < 8; ++r) { mrow[r] = -3.0e38f; lrow[r] = 0.0f; }
  const float scale = 0.125f;   // 1/sqrt(64)

  // prologue: DMA first K/V pair into buffer 0
  tdm_load_2d((unsigned)(unsigned long long)&bufk[0][0], Kb, HDIM, 32, HDIM, SEQ, HDIM);
  tdm_load_2d((unsigned)(unsigned long long)&bufv[0][0], Vb, 32, HDIM, SEQ, HDIM, SEQ);

  for (int jb = 0; jb < SEQ; jb += 32) {
    const int cur = (jb >> 5) & 1;
    const int nxt = cur ^ 1;
    const int jn  = (jb + 32) & (SEQ - 1);   // last prefetch wraps (never read)

    // DMA next pair while we compute the current block
    tdm_load_2d((unsigned)(unsigned long long)&bufk[nxt][0],
                Kb + (size_t)jn * HDIM, HDIM, 32, HDIM, SEQ, HDIM);
    tdm_load_2d((unsigned)(unsigned long long)&bufv[nxt][0],
                Vb + jn, 32, HDIM, SEQ, HDIM, SEQ);
    // 4 outstanding -> wait until the current pair (2 oldest, in-order) landed
    __builtin_amdgcn_s_wait_tensorcnt(2);

    const int msk0 = mk[jb + ln];
    const int msk1 = mk[jb + 16 + ln];

    // ---- scores: S = Q (16x64) . K^T -> two 16x16 tiles, K frags from LDS ----
    const __bf16* kl = &bufk[cur][0];
    v8f s0 = {}, s1 = {};
#pragma unroll
    for (int kk = 0; kk < 2; ++kk) {
      v16bf bk0 = *(const v16bf*)(kl + ln * HDIM + kk * 32 + half * 16);
      v16bf bk1 = *(const v16bf*)(kl + (16 + ln) * HDIM + kk * 32 + half * 16);
      s0 = wmma_bf16(qa[kk], bk0, s0);
      s1 = wmma_bf16(qa[kk], bk1, s1);
    }

    // ---- scale + mask + block row-max (reduce within 16-lane halves) ----
    v8f tmx;
#pragma unroll
    for (int r = 0; r < 8; ++r) {
      s0[r] = msk0 ? s0[r] * scale : -1.0e30f;
      s1[r] = msk1 ? s1[r] * scale : -1.0e30f;
      tmx[r] = fmaxf(s0[r], s1[r]);
    }
#pragma unroll
    for (int d2 = 1; d2 < 16; d2 <<= 1) {
#pragma unroll
      for (int r = 0; r < 8; ++r) tmx[r] = fmaxf(tmx[r], __shfl_xor(tmx[r], d2, 32));
    }

    // ---- online softmax update ----
    v8f fac, rs;
#pragma unroll
    for (int r = 0; r < 8; ++r) {
      const float nm = fmaxf(mrow[r], tmx[r]);
      fac[r]  = __expf(mrow[r] - nm);
      mrow[r] = nm;
      s0[r]   = __expf(s0[r] - nm);
      s1[r]   = __expf(s1[r] - nm);
      rs[r]   = s0[r] + s1[r];
    }
#pragma unroll
    for (int d2 = 1; d2 < 16; d2 <<= 1) {
#pragma unroll
      for (int r = 0; r < 8; ++r) rs[r] += __shfl_xor(rs[r], d2, 32);
    }
#pragma unroll
    for (int r = 0; r < 8; ++r) lrow[r] = lrow[r] * fac[r] + rs[r];
#pragma unroll
    for (int t = 0; t < 4; ++t)
#pragma unroll
      for (int r = 0; r < 8; ++r) o[t][r] *= fac[r];

    // ---- P: C-layout -> LDS -> A-layout fragment ----
#pragma unroll
    for (int r = 0; r < 8; ++r) {
      const int rl = half * 8 + r;
      lds_p[rl * 32 + ln]      = (__bf16)s0[r];
      lds_p[rl * 32 + 16 + ln] = (__bf16)s1[r];
    }
    v16bf pa;
    {
      const __bf16* pp = lds_p + ln * 32 + half * 16;
      v8bf p0 = *(const v8bf*)pp;
      v8bf p1 = *(const v8bf*)(pp + 8);
#pragma unroll
      for (int i = 0; i < 8; ++i) { pa[i] = p0[i]; pa[i + 8] = p1[i]; }
    }

    // ---- O += P (16x32) . V (32x64), V frags from LDS (transposed tile) ----
    const __bf16* vl = &bufv[cur][0];
#pragma unroll
    for (int t = 0; t < 4; ++t) {
      v16bf bv = *(const v16bf*)(vl + (t * 16 + ln) * 32 + half * 16);
      o[t] = wmma_bf16(pa, bv, o[t]);
    }
  }

  // ---- normalize + store attn_vec [b, s, h*64 + d] as bf16 ----
  v8f inv;
#pragma unroll
  for (int r = 0; r < 8; ++r) inv[r] = 1.0f / lrow[r];
#pragma unroll
  for (int t = 0; t < 4; ++t) {
    const int d = t * 16 + ln;
#pragma unroll
    for (int r = 0; r < 8; ++r) {
      const int s = m0 + half * 8 + r;
      av[((size_t)(b * SEQ + s)) * DMODEL + h * HDIM + d] = (__bf16)(o[t][r] * inv[r]);
    }
  }
  // stray wrapped prefetch is retired by S_ENDPGM's implicit wait-idle
}

// ---------------------------------------------------------------------------
extern "C" void kernel_launch(void* const* d_in, const int* in_sizes, int n_in,
                              void* d_out, int out_size, void* d_ws, size_t ws_size,
                              hipStream_t stream) {
  const float* q  = (const float*)d_in[0];
  const float* k  = (const float*)d_in[1];
  const float* v  = (const float*)d_in[2];
  const float* wq = (const float*)d_in[3];
  const float* wk = (const float*)d_in[4];
  const float* wv = (const float*)d_in[5];
  const float* wo = (const float*)d_in[6];
  const int* amask = (const int*)d_in[7];

  char* ws = (char*)d_ws;                         // 32 MiB used
  __bf16* hq  = (__bf16*)(ws);                    //  8 MiB [b,h,s,d]
  __bf16* hk  = (__bf16*)(ws + (8ull  << 20));    //  8 MiB [b,h,s,d]
  __bf16* hvt = (__bf16*)(ws + (16ull << 20));    //  8 MiB [b,h,d,s]
  __bf16* av  = (__bf16*)(ws + (24ull << 20));    //  8 MiB [m, 512]

  dim3 gg(MROWS / 32, DMODEL / 64);               // 256 x 8 waves, 32x64 tiles

  proj_gemm<true, 0><<<gg, 32, 0, stream>>>(q, wq, hq);
  proj_gemm<true, 0><<<gg, 32, 0, stream>>>(k, wk, hk);
  proj_gemm<true, 1><<<gg, 32, 0, stream>>>(v, wv, hvt);

  flash_attn<<<dim3(BSZ * NHEAD, SEQ / 16), 32, 0, stream>>>(hq, hk, hvt, amask, av);

  proj_gemm<false, 2><<<gg, 32, 0, stream>>>(av, wo, d_out);
}
